// torch_Sage_23630910062646
// MI455X (gfx1250) — compile-verified, matrix operand order checked by reading, not run on hardware
//
#include <hip/hip_runtime.h>
#include <hip/hip_bf16.h>
#include <stdint.h>

// ---------------------------------------------------------------------------
// GraphSAGE layer for MI455X (gfx1250, wave32):
//   agg = segment_sum(x[src] * edge_w, dst)        (L2-resident scatter-add)
//   out = relu([agg @ W1 + b1, x @ W2 + b2])       (bf16 WMMA, f32 accumulate)
// ---------------------------------------------------------------------------

#define F_IN   256
#define F_OUT  256
#define LDS_STRIDE 264   // 256 + 8 halfword pad -> conflict-free ds_load_b128

typedef __attribute__((ext_vector_type(16))) __bf16  v16bf;
typedef __attribute__((ext_vector_type(8)))  float   v8f;
typedef __attribute__((ext_vector_type(4)))  unsigned int u32x4;
typedef __attribute__((ext_vector_type(4)))  float   f32x4;

union AFrag {
    v16bf v;
    u32x4 q[2];
};

__device__ __forceinline__ unsigned short f2bf(float f) {
    union { float f; unsigned int u; } c;
    c.f = f;
    unsigned int u = c.u;
    // round-to-nearest-even bf16
    unsigned int r = (u + 0x7FFFu + ((u >> 16) & 1u)) >> 16;
    return (unsigned short)r;
}

// ---------------------------------------------------------------------------
// 0) zero the aggregation buffer (d_ws is poisoned, must init ourselves)
// ---------------------------------------------------------------------------
__global__ void zero_f4(f32x4* __restrict__ p, int n4) {
    int i = blockIdx.x * blockDim.x + threadIdx.x;
    if (i < n4) p[i] = (f32x4){0.f, 0.f, 0.f, 0.f};
}

// ---------------------------------------------------------------------------
// 1) W[k][n] (row-major, [in,out]) -> Wt[n][k] bf16 (per-column contiguous K)
// ---------------------------------------------------------------------------
__global__ void wconv(const float* __restrict__ W, unsigned short* __restrict__ Wt) {
    int k = blockIdx.x;      // 0..255
    int n = threadIdx.x;     // 0..255
    Wt[n * F_IN + k] = f2bf(W[k * F_OUT + n]);
}

// ---------------------------------------------------------------------------
// 2) SpMM scatter: one wave (32 lanes) per edge. x row load is 2 coalesced
//    float4 per lane (1 KB/edge); 8 native f32 atomic adds per lane into agg.
//    x and agg both fit in the 192MB L2 -> atomics resolve at L2 rate.
// ---------------------------------------------------------------------------
__global__ void spmm_edges(const float* __restrict__ x,
                           const int*   __restrict__ src,
                           const int*   __restrict__ dst,
                           const float* __restrict__ ew,
                           float* __restrict__ agg,
                           int E) {
    int e = blockIdx.x * 8 + (threadIdx.x >> 5);   // 8 waves/block, 1 edge/wave
    if (e >= E) return;
    int lane = threadIdx.x & 31;

    int   s = src[e];            // same addr across wave -> broadcast load
    int   d = dst[e];
    float w = ew[e];

    const f32x4* xr = (const f32x4*)(x + (size_t)s * F_IN);
    float*       ar = agg + (size_t)d * F_IN;

    f32x4 v0 = xr[lane];         // cols [4*lane .. 4*lane+3]
    f32x4 v1 = xr[lane + 32];    // cols [128+4*lane .. ]

    int c0 = lane * 4;
    unsafeAtomicAdd(ar + c0 + 0,       v0.x * w);
    unsafeAtomicAdd(ar + c0 + 1,       v0.y * w);
    unsafeAtomicAdd(ar + c0 + 2,       v0.z * w);
    unsafeAtomicAdd(ar + c0 + 3,       v0.w * w);
    unsafeAtomicAdd(ar + c0 + 128,     v1.x * w);
    unsafeAtomicAdd(ar + c0 + 129,     v1.y * w);
    unsafeAtomicAdd(ar + c0 + 130,     v1.z * w);
    unsafeAtomicAdd(ar + c0 + 131,     v1.w * w);
}

// ---------------------------------------------------------------------------
// 3) Fused dual-GEMM + bias + ReLU + concat epilogue.
//    Block = 16 nodes x 512 out cols. 8 waves; each wave runs 2 passes
//    (pass 0: neighbor GEMM -> out[:,0:256], pass 1: self GEMM -> out[:,256:512]).
//    Per pass the wave owns TWO 16-col N-tiles (wave, wave+8) that SHARE the
//    A fragment: half the LDS traffic and two independent WMMA accumulation
//    chains to hide the WMMA->WMMA RAW hazard.
// ---------------------------------------------------------------------------
__global__ void gemm_fused(const float* __restrict__ agg,
                           const float* __restrict__ x,
                           const unsigned short* __restrict__ W1t,
                           const unsigned short* __restrict__ W2t,
                           const float* __restrict__ b1,
                           const float* __restrict__ b2,
                           float* __restrict__ out,
                           int N) {
    __shared__ __align__(16) unsigned short As[2][16][LDS_STRIDE];

    const int m0 = blockIdx.x * 16;
    const int t  = threadIdx.x;

    // Stage 16x256 of agg and x as bf16 into LDS (coalesced f32 reads).
    for (int idx = t; idx < 16 * F_IN; idx += 256) {
        int r = idx >> 8;          // row in tile
        int c = idx & 255;         // column (K)
        int row = m0 + r;
        float va = 0.f, vx = 0.f;
        if (row < N) {
            va = agg[(size_t)row * F_IN + c];
            vx = x  [(size_t)row * F_IN + c];
        }
        As[0][r][c] = f2bf(va);
        As[1][r][c] = f2bf(vx);
    }
    __syncthreads();

    const int wave  = t >> 5;
    const int lane  = t & 31;
    const int l16   = lane & 15;
    const int mrow  = l16;                   // A row index
    const int khalf = (lane >> 4) << 3;      // K offset: lanes 16-31 -> +8
    const bool fullTile = (m0 + 16 <= N);    // block-uniform: no store guards

    #pragma unroll
    for (int which = 0; which < 2; ++which) {   // 0 = neighbor, 1 = self
        const unsigned short* Wt   = which ? W2t : W1t;
        const float*          bias = which ? b2  : b1;

        const int ncol0 = wave * 16 + l16;            // tile "wave"
        const int ncol1 = (wave + 8) * 16 + l16;      // tile "wave+8"
        const unsigned short* wrow0 = Wt + (size_t)ncol0 * F_IN;
        const unsigned short* wrow1 = Wt + (size_t)ncol1 * F_IN;

        v8f acc0 = {};
        v8f acc1 = {};
        #pragma unroll
        for (int kk = 0; kk < F_IN; kk += 32) {
            AFrag a, b0, b1f;
            const unsigned short* ap = &As[which][mrow][kk + khalf];
            a.q[0] = *(const u32x4*)(ap);        // K = kk+khalf .. +7
            a.q[1] = *(const u32x4*)(ap + 16);   // K = kk+khalf+16 .. +23
            const unsigned short* bp0 = wrow0 + kk + khalf;
            const unsigned short* bp1 = wrow1 + kk + khalf;
            b0.q[0]  = *(const u32x4*)(bp0);
            b0.q[1]  = *(const u32x4*)(bp0 + 16);
            b1f.q[0] = *(const u32x4*)(bp1);
            b1f.q[1] = *(const u32x4*)(bp1 + 16);
            // Two independent accumulation chains sharing the A fragment.
            acc0 = __builtin_amdgcn_wmma_f32_16x16x32_bf16(
                       false, a.v, false, b0.v,  (short)0, acc0, false, false);
            acc1 = __builtin_amdgcn_wmma_f32_16x16x32_bf16(
                       false, a.v, false, b1f.v, (short)0, acc1, false, false);
        }

        const float bv0  = bias[ncol0];
        const float bv1  = bias[ncol1];
        const size_t cb  = (size_t)which * F_OUT;     // concat offset
        const int mBase  = m0 + ((lane < 16) ? 0 : 8);

        if (fullTile) {
            #pragma unroll
            for (int i = 0; i < 8; ++i) {
                size_t rowBase = (size_t)(mBase + i) * (2 * F_OUT) + cb;
                float v0 = acc0[i] + bv0;
                float v1 = acc1[i] + bv1;
                out[rowBase + ncol0] = v0 > 0.f ? v0 : 0.f;
                out[rowBase + ncol1] = v1 > 0.f ? v1 : 0.f;
            }
        } else {
            #pragma unroll
            for (int i = 0; i < 8; ++i) {
                int m = mBase + i;
                if (m < N) {
                    size_t rowBase = (size_t)m * (2 * F_OUT) + cb;
                    float v0 = acc0[i] + bv0;
                    float v1 = acc1[i] + bv1;
                    out[rowBase + ncol0] = v0 > 0.f ? v0 : 0.f;
                    out[rowBase + ncol1] = v1 > 0.f ? v1 : 0.f;
                }
            }
        }
    }
}

// ---------------------------------------------------------------------------
// launch
// ---------------------------------------------------------------------------
extern "C" void kernel_launch(void* const* d_in, const int* in_sizes, int n_in,
                              void* d_out, int out_size, void* d_ws, size_t ws_size,
                              hipStream_t stream) {
    const float* x   = (const float*)d_in[0];
    const int*   src = (const int*)  d_in[1];
    const int*   dst = (const int*)  d_in[2];
    const float* ew  = (const float*)d_in[3];
    const float* W1  = (const float*)d_in[4];
    const float* b1  = (const float*)d_in[5];
    const float* W2  = (const float*)d_in[6];
    const float* b2  = (const float*)d_in[7];
    float*       out = (float*)d_out;

    const int N = in_sizes[0] / F_IN;
    const int E = in_sizes[1];

    // Workspace layout: [ agg f32 N*256 | W1t bf16 256*256 | W2t bf16 256*256 ]
    float* agg = (float*)d_ws;
    size_t aggBytes = ((size_t)N * F_IN * sizeof(float) + 255) & ~(size_t)255;
    unsigned short* W1t = (unsigned short*)((char*)d_ws + aggBytes);
    unsigned short* W2t = W1t + F_IN * F_OUT;

    // 0) zero agg
    int n4 = (N * F_IN) / 4;
    zero_f4<<<(n4 + 255) / 256, 256, 0, stream>>>((f32x4*)agg, n4);

    // 1) weights -> bf16, column-major
    wconv<<<F_IN, F_OUT, 0, stream>>>(W1, W1t);
    wconv<<<F_IN, F_OUT, 0, stream>>>(W2, W2t);

    // 2) edge scatter (one wave per edge)
    spmm_edges<<<(E + 7) / 8, 256, 0, stream>>>(x, src, dst, ew, agg, E);

    // 3) fused dual-GEMM + bias + relu + concat
    gemm_fused<<<(N + 15) / 16, 256, 0, stream>>>(agg, x, W1t, W2t, b1, b2, out, N);
}
